// ConvexMLPBlock_15925738734246
// MI455X (gfx1250) — matrix operand order repaired and checked by reading, not compiled, not run on hardware
//
#include <hip/hip_runtime.h>
#include <hip/hip_bf16.h>

// ---- problem constants (from reference) ----
#define B_   64
#define HW_  196
#define HWP_ 224      // HW padded to multiple of 32 (WMMA K-step)
#define D_   768
#define E_   256
#define C_   10

typedef __attribute__((ext_vector_type(16))) __bf16 v16bf;
typedef __attribute__((ext_vector_type(8)))  __bf16 v8bf;
typedef __attribute__((ext_vector_type(8)))  float  v8f;

// Load the 16-bit A fragment for this lane: two contiguous 8-elem (16B) runs.
// row = lane's matrix row; per ISA 7.12.2: lanes 0-15 hold K {0..7,16..23},
// lanes 16-31 hold K {8..15,24..31} (relative to k0).
__device__ __forceinline__ v16bf load_a_frag(const __bf16* row, int k0, bool hi) {
    const int base = k0 + (hi ? 8 : 0);
    v8bf lo = *(const v8bf*)(row + base);
    v8bf up = *(const v8bf*)(row + base + 16);
    return __builtin_shufflevector(lo, up, 0,1,2,3,4,5,6,7,8,9,10,11,12,13,14,15);
}

// Load the 16-bit B fragment: two contiguous 8-elem (16B) runs of the column
// (stored as a row of the transposed operand). Lanes 0-15: K {0..7,16..23}; 16-31: K {8..15,24..31}.
__device__ __forceinline__ v16bf load_b_frag(const __bf16* col, int k0, bool hi) {
    const int base = k0 + (hi ? 8 : 0);
    v8bf lo = *(const v8bf*)(col + base);
    v8bf up = *(const v8bf*)(col + base + 16);
    return __builtin_shufflevector(lo, up, 0,1,2,3,4,5,6,7,8,9,10,11,12,13,14,15);
}

// ---------------------------------------------------------------------------
// Pack 1: x (fp32, BxHWxD) -> xbf (bf16 row-major) and xT (bf16, BxDxHWP,
// zero-padded in h). LDS 32x32 tile transpose; reads and writes coalesced.
// ---------------------------------------------------------------------------
__global__ __launch_bounds__(256)
void pack_x_kernel(const float* __restrict__ x,
                   __bf16* __restrict__ xbf,
                   __bf16* __restrict__ xT) {
    __shared__ __bf16 tile[32][33];
    const int b  = blockIdx.z;
    const int h0 = blockIdx.y * 32;     // 7 tiles cover 0..223
    const int d0 = blockIdx.x * 32;     // 24 tiles
    const int tx = threadIdx.x & 31;
    const int ty = threadIdx.x >> 5;    // 0..7

    const float* xb = x + (size_t)b * HW_ * D_;
#pragma unroll
    for (int r = ty; r < 32; r += 8) {
        const int h = h0 + r;
        const float v = (h < HW_) ? xb[(size_t)h * D_ + d0 + tx] : 0.0f;
        const __bf16 bv = (__bf16)v;
        tile[r][tx] = bv;
        if (h < HW_)
            xbf[((size_t)b * HW_ + h) * D_ + d0 + tx] = bv;
    }
    __syncthreads();
#pragma unroll
    for (int r = ty; r < 32; r += 8)
        xT[((size_t)b * D_ + d0 + r) * HWP_ + h0 + tx] = tile[tx][r];
}

// Pack 2: ag_w fp32 -> bf16
__global__ __launch_bounds__(256)
void pack_w_kernel(const float* __restrict__ w, __bf16* __restrict__ wbf, int n) {
    const int i = blockIdx.x * 256 + threadIdx.x;
    if (i < n) wbf[i] = (__bf16)w[i];
}

// ---------------------------------------------------------------------------
// Kernel 1: G = x[b] @ ag_w^T + ag_b ; write sign pattern TRANSPOSED and
// zero-padded as bf16 0/1 into St[b][e][hp]. One wave per 16x16 tile.
// ---------------------------------------------------------------------------
__global__ __launch_bounds__(256)
void sign_gemm_kernel(const __bf16* __restrict__ xbf,
                      const __bf16* __restrict__ wbf,
                      const float* __restrict__ ag_b,
                      __bf16* __restrict__ St) {
    const int lane = threadIdx.x & 31;
    const int wave = threadIdx.x >> 5;
    const bool hi  = lane >= 16;
    const int l16  = lane & 15;

    const int h0 = blockIdx.x * 16;                 // 14 tiles (h up to 223)
    const int e0 = (blockIdx.y * 8 + wave) * 16;    // 16 e-tiles
    const int b  = blockIdx.z;

    // A row: clamp OOB rows (their D rows are discarded at store time)
    int am = h0 + l16; if (am >= HW_) am = HW_ - 1;
    const __bf16* arow = xbf + ((size_t)b * HW_ + am) * D_;
    const __bf16* brow = wbf + (size_t)(e0 + l16) * D_;

    v8f acc = {};
    for (int k0 = 0; k0 < D_; k0 += 32) {           // 24 WMMA iterations
        const v16bf a  = load_a_frag(arow, k0, hi);
        const v16bf bm = load_b_frag(brow, k0, hi);
        acc = __builtin_amdgcn_wmma_f32_16x16x32_bf16(
            false, a, false, bm, (short)0, acc, false, false);
    }

    // Lane's 8 C values share column e = e0+l16; rows h0 + r + (hi?8:0).
    const float bias = ag_b[e0 + l16];
    __bf16* srow = St + ((size_t)b * E_ + (e0 + l16)) * HWP_;
    const int mbase = h0 + (hi ? 8 : 0);
#pragma unroll
    for (int r = 0; r < 8; ++r) {
        const int h = mbase + r;                    // < 224 always
        srow[h] = (h < HW_ && acc[r] + bias > 0.0f) ? (__bf16)1.0f : (__bf16)0.0f;
    }
}

// ---------------------------------------------------------------------------
// Kernel 2: M = St[b] (256x224) @ x[b] (224x768, via xT), then contract each
// 16x16 M-tile with lm_w over the 10 classes, wave-reduce, atomicAdd.
// ---------------------------------------------------------------------------
__global__ __launch_bounds__(256)
void masked_reduce_kernel(const __bf16* __restrict__ St,
                          const __bf16* __restrict__ xT,
                          const float* __restrict__ lm_w,
                          float* __restrict__ out) {
    const int lane = threadIdx.x & 31;
    const int wave = threadIdx.x >> 5;
    const bool hi  = lane >= 16;
    const int l16  = lane & 15;

    const int d0 = blockIdx.x * 16;                 // 48 d-tiles
    const int e0 = (blockIdx.y * 8 + wave) * 16;    // 16 e-tiles
    const int b  = blockIdx.z;

    const __bf16* srow = St + ((size_t)b * E_ + (e0 + l16)) * HWP_;
    const __bf16* xrow = xT + ((size_t)b * D_ + (d0 + l16)) * HWP_;
    const int dcol = d0 + l16;

    // Warm L2 for the lm_w rows this lane will contract against.
    __builtin_prefetch(lm_w + (size_t)(e0 + l16) * C_ * D_ + dcol, 0, 1);

    v8f acc = {};
#pragma unroll
    for (int k0 = 0; k0 < HWP_; k0 += 32) {         // 7 WMMA iterations, guard-free
        const v16bf a  = load_a_frag(srow, k0, hi);
        const v16bf bm = load_b_frag(xrow, k0, hi);
        acc = __builtin_amdgcn_wmma_f32_16x16x32_bf16(
            false, a, false, bm, (short)0, acc, false, false);
    }

    // acc[r] = M[e0 + r + (hi?8:0)][dcol]; contract with lm_w[(e*C+c)*D + d]
    float partial[C_];
#pragma unroll
    for (int c = 0; c < C_; ++c) partial[c] = 0.0f;

#pragma unroll
    for (int r = 0; r < 8; ++r) {
        const int e = e0 + r + (hi ? 8 : 0);
        const float mv = acc[r];
        const float* lrow = lm_w + (size_t)e * C_ * D_ + dcol;
#pragma unroll
        for (int c = 0; c < C_; ++c)
            partial[c] += mv * lrow[(size_t)c * D_];
    }

    const float scale = 1.0f / ((float)HW_ * (float)E_);
#pragma unroll
    for (int c = 0; c < C_; ++c) {
        float v = partial[c];
#pragma unroll
        for (int off = 16; off > 0; off >>= 1)
            v += __shfl_xor(v, off, 32);
        if (lane == 0)
            atomicAdd(&out[b * C_ + c], v * scale);
    }
}

// ---------------------------------------------------------------------------
extern "C" void kernel_launch(void* const* d_in, const int* in_sizes, int n_in,
                              void* d_out, int out_size, void* d_ws, size_t ws_size,
                              hipStream_t stream) {
    const float* x    = (const float*)d_in[0];   // (64, 196, 768)
    const float* ag_w = (const float*)d_in[1];   // (256, 768)
    const float* ag_b = (const float*)d_in[2];   // (256,)
    const float* lm_w = (const float*)d_in[3];   // (2560, 768)
    float* out = (float*)d_out;                  // (64, 10)

    // Workspace layout (256B-aligned segments), total ~48.6 MB:
    char* ws = (char*)d_ws;
    const size_t xbf_bytes = (size_t)B_ * HW_  * D_   * 2;   // 19,267,584
    const size_t xT_bytes  = (size_t)B_ * D_   * HWP_ * 2;   // 22,020,096
    __bf16* xbf = (__bf16*)ws;
    __bf16* xT  = (__bf16*)(ws + ((xbf_bytes + 255) & ~(size_t)255));
    __bf16* St  = (__bf16*)(ws + ((xbf_bytes + 255) & ~(size_t)255)
                               + ((xT_bytes  + 255) & ~(size_t)255));
    __bf16* wbf = St + (size_t)B_ * E_ * HWP_;               // + 7,340,032 (+0.4MB)

    hipMemsetAsync(d_out, 0, (size_t)out_size * sizeof(float), stream);

    dim3 blk(256);
    pack_x_kernel<<<dim3(D_ / 32, HWP_ / 32, B_), blk, 0, stream>>>(x, xbf, xT);
    pack_w_kernel<<<dim3((E_ * D_ + 255) / 256), blk, 0, stream>>>(ag_w, wbf, E_ * D_);

    sign_gemm_kernel<<<dim3(HWP_ / 16, E_ / (16 * 8), B_), blk, 0, stream>>>(
        xbf, wbf, ag_b, St);

    masked_reduce_kernel<<<dim3(D_ / 16, E_ / (16 * 8), B_), blk, 0, stream>>>(
        St, xT, lm_w, out);
}
